// PointNet2SSGClassification_18837726561045
// MI455X (gfx1250) — matrix operand order repaired and checked by reading, not compiled, not run on hardware
//
#include <hip/hip_runtime.h>
#include <hip/hip_bf16.h>

typedef __attribute__((ext_vector_type(16))) _Float16 v16h;
typedef __attribute__((ext_vector_type(8)))  float    v8f;
typedef __attribute__((ext_vector_type(4)))  int      v4i;

#define BATCH 16
#define NPTS  4096

// Async global->LDS (CDNA5 GLOBAL_LOAD_ASYNC_TO_LDS_B128, ASYNCcnt-tracked)
#if defined(__has_builtin)
#  if __has_builtin(__builtin_amdgcn_global_load_async_to_lds_b128) && \
      __has_builtin(__builtin_amdgcn_s_wait_asynccnt)
#    define HAVE_ASYNC 1
#  endif
#endif
#ifndef HAVE_ASYNC
#  define HAVE_ASYNC 0
#endif

#if HAVE_ASYNC
typedef __attribute__((address_space(1))) v4i* as1_v4i;
typedef __attribute__((address_space(3))) v4i* as3_v4i;
#endif

constexpr int A_F16 = 0;   // A source: LDS f16
constexpr int A_F32 = 1;   // A source: f32 (LDS or global by pointer provenance)
constexpr int O_LDS_F16  = 0;
constexpr int O_GLB_F32  = 1;
constexpr int O_MAXSTAGE = 2;

// ---------------------------------------------------------------------------
// WMMA fragment loaders (wave32, v_wmma_f32_16x16x32_f16)
// A: M x K row-major. lane l: row = m0+(l&15); halves e<8 -> K=k0+grp*8+e,
//    e>=8 -> K=k0+16+grp*8+(e-8), grp = (l>=16)*8.
// B: K x N row-major. lane l -> N=n0+(l&15); halves e -> K as for A.
// C/D: lane l, vgpr v -> row = m0+(l>=16)*8+v, col = n0+(l&15).
// ---------------------------------------------------------------------------
static __device__ inline v16h load_a_f16(const _Float16* A, int ld, int m0, int k0) {
  int lane = threadIdx.x & 31;
  int grp = (lane >> 4) << 3;
  const _Float16* row = A + (size_t)(m0 + (lane & 15)) * ld + k0;
  v16h r;
#pragma unroll
  for (int e = 0; e < 8; ++e) r[e] = row[grp + e];
#pragma unroll
  for (int e = 0; e < 8; ++e) r[8 + e] = row[16 + grp + e];
  return r;
}

static __device__ inline v16h load_a_f32(const float* A, int ld, int m0, int k0) {
  int lane = threadIdx.x & 31;
  int grp = (lane >> 4) << 3;
  const float* row = A + (size_t)(m0 + (lane & 15)) * ld + k0;
  v16h r;
#pragma unroll
  for (int e = 0; e < 8; ++e) r[e] = (_Float16)row[grp + e];
#pragma unroll
  for (int e = 0; e < 8; ++e) r[8 + e] = (_Float16)row[16 + grp + e];
  return r;
}

static __device__ inline v16h load_b_lds(const _Float16* W, int ldw, int k0, int n0) {
  int lane = threadIdx.x & 31;
  int grp = (lane >> 4) << 3;
  const _Float16* row = W + (size_t)k0 * ldw + n0 + (lane & 15);
  v16h r;
#pragma unroll
  for (int e = 0; e < 8; ++e) r[e] = row[(size_t)(grp + e) * ldw];
#pragma unroll
  for (int e = 0; e < 8; ++e) r[8 + e] = row[(size_t)(16 + grp + e) * ldw];
  return r;
}

static __device__ inline v16h load_b_glb(const float* W, int ldw, int k0, int n0) {
  int lane = threadIdx.x & 31;
  int grp = (lane >> 4) << 3;
  const float* row = W + (size_t)k0 * ldw + n0 + (lane & 15);
  __builtin_prefetch(row + (size_t)32 * ldw, 0, 1);  // next K chunk
  v16h r;
#pragma unroll
  for (int e = 0; e < 8; ++e) r[e] = (_Float16)row[(size_t)(grp + e) * ldw];
#pragma unroll
  for (int e = 0; e < 8; ++e) r[8 + e] = (_Float16)row[(size_t)(16 + grp + e) * ldw];
  return r;
}

// ---------------------------------------------------------------------------
// Tiled MLP layer: out = relu(actIn @ W + bias), M = MT*16 rows.
// Compile-time specialized on A-source (AM), W-source (WLDS), output (OM).
// K = 32*KC WMMA chunks + Ktail VALU rows folded into the accumulator init
// (tail weights come from wTail rows 0..Ktail-1, global f32).
// All wave loops are uniform -> EXEC all-ones around v_wmma (ISA 7.12 req).
// ---------------------------------------------------------------------------
template <int AM, bool WLDS, int OM>
static __device__ void mlp_tiles(
    const void* aP, int lda,
    const void* wP, int ldw,
    const float* wTail, const float* bias,
    int MT, int NTbeg, int NTend, int KC, int Ktail,
    void* outP, int ldo,
    float* stage, int stageLd, int nwaves)
{
  const _Float16* aH = (const _Float16*)aP;
  const float*    aF = (const float*)aP;
  const _Float16* wL = (const _Float16*)wP;
  const float*    wG = (const float*)wP;
  int wave = threadIdx.x >> 5;
  int lane = threadIdx.x & 31;
  int grp = lane >> 4;
  int col = lane & 15;
  int total = MT * (NTend - NTbeg);
  for (int t = wave; t < total; t += nwaves) {
    int mt = t % MT;
    int nt = NTbeg + t / MT;
    int m0 = mt * 16, n0 = nt * 16;
    float binit = bias[n0 + col];
    v8f acc;
#pragma unroll
    for (int v = 0; v < 8; ++v) acc[v] = binit;
    if (Ktail > 0) {
      int kb = KC * 32;
#pragma unroll
      for (int v = 0; v < 8; ++v) {
        int row = m0 + grp * 8 + v;
        float s = 0.f;
        for (int k = 0; k < Ktail; ++k) {
          float a = (AM == A_F16) ? (float)aH[(size_t)row * lda + kb + k]
                                  : aF[(size_t)row * lda + kb + k];
          s += a * wTail[(size_t)k * ldw + n0 + col];
        }
        acc[v] += s;
      }
    }
    for (int kc = 0; kc < KC; ++kc) {
      int k0 = kc * 32;
      v16h af = (AM == A_F16) ? load_a_f16(aH, lda, m0, k0)
                              : load_a_f32(aF, lda, m0, k0);
      v16h bf = WLDS ? load_b_lds(wL, ldw, k0, n0)
                     : load_b_glb(wG, ldw, k0, n0);
      acc = __builtin_amdgcn_wmma_f32_16x16x32_f16(
          false, af, false, bf, (short)0, acc, false, false);
    }
    if (OM == O_MAXSTAGE) {
      float m = 0.f;                      // relu floor
#pragma unroll
      for (int v = 0; v < 8; ++v) { float r = acc[v]; if (r > m) m = r; }
      float o = __shfl_xor(m, 16);        // lane-group swap -> max over 16 rows
      if (o > m) m = o;
      if (grp == 0) stage[mt * stageLd + (n0 - NTbeg * 16) + col] = m;
    } else if (OM == O_LDS_F16) {
      _Float16* out = (_Float16*)outP;
#pragma unroll
      for (int v = 0; v < 8; ++v) {
        float r = acc[v]; r = r > 0.f ? r : 0.f;
        out[(size_t)(m0 + grp * 8 + v) * ldo + n0 + col] = (_Float16)r;
      }
    } else {
      float* out = (float*)outP;
#pragma unroll
      for (int v = 0; v < 8; ++v) {
        float r = acc[v]; r = r > 0.f ? r : 0.f;
        out[(size_t)(m0 + grp * 8 + v) * ldo + n0 + col] = r;
      }
    }
  }
}

// ---------------------------------------------------------------------------
// (B,3,N) -> (B,N,3)
// ---------------------------------------------------------------------------
__global__ void transpose_kernel(const float* __restrict__ in, float* __restrict__ xyz) {
  int g = blockIdx.x * blockDim.x + threadIdx.x;
  if (g >= BATCH * NPTS) return;
  int b = g / NPTS, i = g % NPTS;
  xyz[(size_t)g * 3 + 0] = in[(size_t)(b * 3 + 0) * NPTS + i];
  xyz[(size_t)g * 3 + 1] = in[(size_t)(b * 3 + 1) * NPTS + i];
  xyz[(size_t)g * 3 + 2] = in[(size_t)(b * 3 + 2) * NPTS + i];
}

// ---------------------------------------------------------------------------
// Farthest point sampling. One block per batch, 512 threads, points in regs.
// Matches jnp scan: idx[it] = previous far; argmax ties -> lowest index.
// ---------------------------------------------------------------------------
__global__ __launch_bounds__(512)
void fps_kernel(const float* __restrict__ xyz, int n, int npoint,
                int* __restrict__ fidx, float* __restrict__ newxyz)
{
  __shared__ float sval[512];
  __shared__ int   sidx[512];
  __shared__ float scent[3];
  __shared__ int   sfar;
  int b = blockIdx.x, tid = threadIdx.x;
  const float* P = xyz + (size_t)b * n * 3;
  float px[8], py[8], pz[8], dist[8];
  int ppt = n >> 9;
  for (int j = 0; j < ppt; ++j) {
    int i = tid + j * 512;
    px[j] = P[i * 3 + 0]; py[j] = P[i * 3 + 1]; pz[j] = P[i * 3 + 2];
    dist[j] = 1e10f;
  }
  if (tid == 0) sfar = 0;
  __syncthreads();
  for (int it = 0; it < npoint; ++it) {
    if (tid == 0) {
      int far = sfar;
      fidx[b * npoint + it] = far;
      scent[0] = P[far * 3 + 0]; scent[1] = P[far * 3 + 1]; scent[2] = P[far * 3 + 2];
      float* NX = newxyz + (size_t)(b * npoint + it) * 3;
      NX[0] = scent[0]; NX[1] = scent[1]; NX[2] = scent[2];
    }
    __syncthreads();
    float cx = scent[0], cy = scent[1], cz = scent[2];
    float best = -1.f; int bi = n;
    for (int j = 0; j < ppt; ++j) {
      float dx = px[j] - cx, dy = py[j] - cy, dz = pz[j] - cz;
      float d = dx * dx + dy * dy + dz * dz;
      if (d < dist[j]) dist[j] = d;
      int i = tid + j * 512;
      if (dist[j] > best || (dist[j] == best && i < bi)) { best = dist[j]; bi = i; }
    }
    sval[tid] = best; sidx[tid] = bi;
    __syncthreads();
    for (int off = 256; off > 0; off >>= 1) {
      if (tid < off) {
        float v2 = sval[tid + off]; int i2 = sidx[tid + off];
        if (v2 > sval[tid] || (v2 == sval[tid] && i2 < sidx[tid])) {
          sval[tid] = v2; sidx[tid] = i2;
        }
      }
      __syncthreads();
    }
    if (tid == 0) sfar = sidx[0];
    __syncthreads();
  }
}

// ---------------------------------------------------------------------------
// Ball query: ordered scan == sort + take-first-nsample of the reference.
// ---------------------------------------------------------------------------
__global__ void ballq_kernel(const float* __restrict__ xyz,
                             const float* __restrict__ newxyz,
                             int n, int s, float r2, int nsample,
                             int* __restrict__ out, int total)
{
  int g = blockIdx.x * blockDim.x + threadIdx.x;
  if (g >= total) return;
  int b = g / s;
  const float* P = xyz + (size_t)b * n * 3;
  const float* C = newxyz + (size_t)g * 3;
  float cx = C[0], cy = C[1], cz = C[2];
  int* O = out + (size_t)g * nsample;
  int cnt = 0, first = -1;
  for (int i = 0; i < n && cnt < nsample; ++i) {
    float dx = P[i * 3 + 0] - cx, dy = P[i * 3 + 1] - cy, dz = P[i * 3 + 2] - cz;
    float d = dx * dx + dy * dy + dz * dz;
    if (d <= r2) { if (first < 0) first = i; O[cnt++] = i; }
  }
  int fill = first >= 0 ? first : n - 1;
  for (; cnt < nsample; ++cnt) O[cnt] = fill;
}

// ---------------------------------------------------------------------------
// SA1: 4 centroids/WG (128 rows), 3->64 (VALU) ->64->128 (WMMA, LDS weights),
// fused max over 32 samples. grid = B*512/4, block 128.  LDS ~60 KB.
// ---------------------------------------------------------------------------
__global__ __launch_bounds__(128)
void sa1_kernel(const float* __restrict__ xyz, const float* __restrict__ newxyz,
                const int* __restrict__ idx,
                const float* __restrict__ w1, const float* __restrict__ b1,
                const float* __restrict__ w2, const float* __restrict__ b2,
                const float* __restrict__ w3, const float* __restrict__ b3,
                float* __restrict__ l1f)
{
  __shared__ _Float16 sw2[64 * 64];
  __shared__ _Float16 sw3[64 * 128];
  __shared__ _Float16 act1[128 * 64];
  __shared__ _Float16 act2[128 * 64];
  __shared__ float    stage[8 * 128];
  int tid = threadIdx.x;
  int cent0 = blockIdx.x * 4;
  int bIdx = cent0 >> 9;
  for (int i = tid; i < 64 * 64; i += 128)  sw2[i] = (_Float16)w2[i];
  for (int i = tid; i < 64 * 128; i += 128) sw3[i] = (_Float16)w3[i];
  {
    int r = tid;
    int c = r >> 5, s = r & 31;
    int cg = cent0 + c;
    int pi = idx[(size_t)cg * 32 + s];
    const float* P = xyz + (size_t)(bIdx * NPTS + pi) * 3;
    const float* C = newxyz + (size_t)cg * 3;
    float g0 = P[0] - C[0], g1 = P[1] - C[1], g2 = P[2] - C[2];
    for (int o = 0; o < 64; ++o) {
      float v = b1[o] + g0 * w1[o] + g1 * w1[64 + o] + g2 * w1[128 + o];
      act1[r * 64 + o] = (_Float16)(v > 0.f ? v : 0.f);
    }
  }
  __syncthreads();
  mlp_tiles<A_F16, true, O_LDS_F16>(act1, 64, sw2, 64, nullptr, b2,
                                    8, 0, 4, 2, 0, act2, 64, nullptr, 0, 4);
  __syncthreads();
  mlp_tiles<A_F16, true, O_MAXSTAGE>(act2, 64, sw3, 128, nullptr, b3,
                                     8, 0, 8, 2, 0, nullptr, 0, stage, 128, 4);
  __syncthreads();
  for (int i = tid; i < 4 * 128; i += 128) {
    int c = i >> 7, o = i & 127;
    float a = stage[(2 * c) * 128 + o], bb = stage[(2 * c + 1) * 128 + o];
    l1f[(size_t)(cent0 + c) * 128 + o] = a > bb ? a : bb;
  }
}

// ---------------------------------------------------------------------------
// SA2: 1 centroid/WG (64 rows = 4 M-tiles), K = 131 = 4 WMMA chunks (feats,
// weight rows 3..130) + 3-row VALU tail (xyz, weight rows 0..2).
// Feature rows pulled with GLOBAL_LOAD_ASYNC_TO_LDS_B128 when available.
// Layer-2 output aliases the f32 act0 buffer.  grid = B*128, block 128.
// LDS: 33792 + 16384 + 2048 = ~51 KB.
// ---------------------------------------------------------------------------
__global__ __launch_bounds__(128)
void sa2_kernel(const float* __restrict__ l1xyz, const float* __restrict__ l1f,
                const float* __restrict__ newxyz, const int* __restrict__ idx,
                const float* __restrict__ w1, const float* __restrict__ b1,
                const float* __restrict__ w2, const float* __restrict__ b2,
                const float* __restrict__ w3, const float* __restrict__ b3,
                float* __restrict__ l2f)
{
  __shared__ float    act0f[64 * 132];       // feats cols 0..127, xyz 128..130
  __shared__ _Float16 act1[64 * 128];
  __shared__ float    stage[4 * 128];
  _Float16* act2 = reinterpret_cast<_Float16*>(act0f);   // reused after layer1
  int tid = threadIdx.x;
  int cent = blockIdx.x;
  int b = cent >> 7;
#if HAVE_ASYNC
  {
    int r = tid >> 5;                        // 4 rows per pass, 32 lanes each
    int lane = tid & 31;
    for (int base = 0; base < 64; base += 4) {
      int rr = base + r;
      int pi = idx[(size_t)cent * 64 + rr];
      const float* src = l1f + (size_t)(b * 512 + pi) * 128 + lane * 4;
      float* dst = act0f + rr * 132 + lane * 4;   // 16B-aligned (132*4=528=33*16)
      __builtin_amdgcn_global_load_async_to_lds_b128(
          (as1_v4i)src, (as3_v4i)dst, 0, 0);
    }
  }
#else
  for (int i = tid; i < 64 * 128; i += 128) {
    int r = i >> 7, k = i & 127;
    int pi = idx[(size_t)cent * 64 + r];
    act0f[r * 132 + k] = l1f[(size_t)(b * 512 + pi) * 128 + k];
  }
#endif
  for (int i = tid; i < 64 * 3; i += 128) {
    int r = i / 3, k = i % 3;
    int pi = idx[(size_t)cent * 64 + r];
    act0f[r * 132 + 128 + k] =
        l1xyz[(size_t)(b * 512 + pi) * 3 + k] - newxyz[(size_t)cent * 3 + k];
  }
#if HAVE_ASYNC
  __builtin_amdgcn_s_wait_asynccnt(0);
#endif
  __syncthreads();
  mlp_tiles<A_F32, false, O_LDS_F16>(act0f, 132, w1 + 3 * 128, 128, w1, b1,
                                     4, 0, 8, 4, 3, act1, 128, nullptr, 0, 4);
  __syncthreads();
  mlp_tiles<A_F16, false, O_LDS_F16>(act1, 128, w2, 128, nullptr, b2,
                                     4, 0, 8, 4, 0, act2, 128, nullptr, 0, 4);
  __syncthreads();
  mlp_tiles<A_F16, false, O_MAXSTAGE>(act2, 128, w3, 128, nullptr, b3,
                                      4, 0, 8, 4, 0, nullptr, 0, stage, 128, 4);
  __syncthreads();
  for (int o = tid; o < 128; o += 128) {
    float m = stage[o];
    for (int t = 1; t < 4; ++t) { float v = stage[t * 128 + o]; if (v > m) m = v; }
    l2f[(size_t)cent * 128 + o] = m;
  }
}

// ---------------------------------------------------------------------------
// SA3 act0 build: [B][128][132] f32, cols 0..127 = l2_f, 128..130 = l2_xyz.
// ---------------------------------------------------------------------------
__global__ void sa3_act0_kernel(const float* __restrict__ l2xyz,
                                const float* __restrict__ l2f,
                                float* __restrict__ act0)
{
  int i = blockIdx.x * blockDim.x + threadIdx.x;
  if (i >= BATCH * 128 * 131) return;
  int k = i % 131, rg = i / 131;
  float v = (k < 128) ? l2f[(size_t)rg * 128 + k] : l2xyz[(size_t)rg * 3 + (k - 128)];
  act0[(size_t)rg * 132 + k] = v;
}

// ---------------------------------------------------------------------------
// Global<->global WMMA GEMM layer (SA3 + decoder). Per WG: one batch, all MT
// M-tiles, ntPer N-tiles. OM = O_GLB_F32 (store relu act) or O_MAXSTAGE
// (fused max-pool over M into out[b][N]).
// ---------------------------------------------------------------------------
template <int OM>
__global__ __launch_bounds__(128)
void gemm_kernel(const float* __restrict__ A, int MT, int lda,
                 const float* __restrict__ W, int ldw,
                 const float* __restrict__ wTail, const float* __restrict__ bias,
                 int KC, int Ktail, int ntPer, int Ntotal,
                 float* __restrict__ out, int ldo)
{
  __shared__ float stage[8 * 128];
  int nTiles = Ntotal >> 4;
  int ng = nTiles / ntPer;
  int b = blockIdx.x / ng, g = blockIdx.x % ng;
  int NTbeg = g * ntPer, NTend = NTbeg + ntPer;
  const float* Ab = A + (size_t)b * (MT * 16) * lda;
  if (OM == O_GLB_F32) {
    mlp_tiles<A_F32, false, O_GLB_F32>(Ab, lda, W, ldw, wTail, bias,
                                       MT, NTbeg, NTend, KC, Ktail,
                                       out + (size_t)b * (MT * 16) * ldo, ldo,
                                       nullptr, 0, 4);
  } else {
    int cols = ntPer * 16;
    mlp_tiles<A_F32, false, O_MAXSTAGE>(Ab, lda, W, ldw, wTail, bias,
                                        MT, NTbeg, NTend, KC, Ktail,
                                        nullptr, 0, stage, cols, 4);
    __syncthreads();
    for (int i = threadIdx.x; i < cols; i += 128) {
      float m = stage[i];
      for (int t = 1; t < MT; ++t) { float v = stage[t * cols + i]; if (v > m) m = v; }
      out[(size_t)b * Ntotal + NTbeg * 16 + i] = m;
    }
  }
}

// ---------------------------------------------------------------------------
// Decoder last layer 256 -> 40, no relu.
// ---------------------------------------------------------------------------
__global__ void dec_final_kernel(const float* __restrict__ a2,
                                 const float* __restrict__ w,
                                 const float* __restrict__ b,
                                 float* __restrict__ out)
{
  int i = blockIdx.x * blockDim.x + threadIdx.x;
  if (i >= 16 * 40) return;
  int r = i / 40, o = i % 40;
  float s = b[o];
  for (int k = 0; k < 256; ++k) s += a2[r * 256 + k] * w[k * 40 + o];
  out[i] = s;
}

// ---------------------------------------------------------------------------
extern "C" void kernel_launch(void* const* d_in, const int* in_sizes, int n_in,
                              void* d_out, int out_size, void* d_ws, size_t ws_size,
                              hipStream_t stream) {
  const float* inputs = (const float*)d_in[0];
  const float* sa1w[3] = {(const float*)d_in[1], (const float*)d_in[2], (const float*)d_in[3]};
  const float* sa1b[3] = {(const float*)d_in[4], (const float*)d_in[5], (const float*)d_in[6]};
  const float* sa2w[3] = {(const float*)d_in[7], (const float*)d_in[8], (const float*)d_in[9]};
  const float* sa2b[3] = {(const float*)d_in[10], (const float*)d_in[11], (const float*)d_in[12]};
  const float* sa3w[3] = {(const float*)d_in[13], (const float*)d_in[14], (const float*)d_in[15]};
  const float* sa3b[3] = {(const float*)d_in[16], (const float*)d_in[17], (const float*)d_in[18]};
  const float* decw[3] = {(const float*)d_in[19], (const float*)d_in[20], (const float*)d_in[21]};
  const float* decb[3] = {(const float*)d_in[22], (const float*)d_in[23], (const float*)d_in[24]};

  float* base = (float*)d_ws;
  size_t off = 0;
  auto allocF = [&](size_t n) { float* r = base + off; off += n; return r; };
  float* xyz    = allocF((size_t)BATCH * NPTS * 3);
  float* l1xyz  = allocF((size_t)BATCH * 512 * 3);
  float* l2xyz  = allocF((size_t)BATCH * 128 * 3);
  float* l1f    = allocF((size_t)BATCH * 512 * 128);
  float* l2f    = allocF((size_t)BATCH * 128 * 128);
  int*   fidx1  = (int*)allocF((size_t)BATCH * 512);
  int*   fidx2  = (int*)allocF((size_t)BATCH * 128);
  int*   idx1   = (int*)allocF((size_t)BATCH * 512 * 32);
  int*   idx2   = (int*)allocF((size_t)BATCH * 128 * 64);
  float* a0     = allocF((size_t)BATCH * 128 * 132);
  float* a1     = allocF((size_t)BATCH * 128 * 256);
  float* a2     = allocF((size_t)BATCH * 128 * 512);
  float* s3out  = allocF((size_t)BATCH * 1024);
  float* dact1  = allocF((size_t)16 * 512);
  float* dact2  = allocF((size_t)16 * 256);
  (void)ws_size; (void)in_sizes; (void)n_in; (void)out_size;

  transpose_kernel<<<(BATCH * NPTS + 255) / 256, 256, 0, stream>>>(inputs, xyz);

  // ---- SA1 ----
  fps_kernel<<<BATCH, 512, 0, stream>>>(xyz, NPTS, 512, fidx1, l1xyz);
  ballq_kernel<<<(BATCH * 512 + 255) / 256, 256, 0, stream>>>(
      xyz, l1xyz, NPTS, 512, 0.2f * 0.2f, 32, idx1, BATCH * 512);
  sa1_kernel<<<BATCH * 512 / 4, 128, 0, stream>>>(
      xyz, l1xyz, idx1, sa1w[0], sa1b[0], sa1w[1], sa1b[1], sa1w[2], sa1b[2], l1f);

  // ---- SA2 ----
  fps_kernel<<<BATCH, 512, 0, stream>>>(l1xyz, 512, 128, fidx2, l2xyz);
  ballq_kernel<<<(BATCH * 128 + 255) / 256, 256, 0, stream>>>(
      l1xyz, l2xyz, 512, 128, 0.4f * 0.4f, 64, idx2, BATCH * 128);
  sa2_kernel<<<BATCH * 128, 128, 0, stream>>>(
      l1xyz, l1f, l2xyz, idx2, sa2w[0], sa2b[0], sa2w[1], sa2b[1], sa2w[2], sa2b[2], l2f);

  // ---- SA3 (group-all): 131 -> 256 -> 512 -> 1024, max over 128 pts ----
  sa3_act0_kernel<<<(BATCH * 128 * 131 + 255) / 256, 256, 0, stream>>>(l2xyz, l2f, a0);
  gemm_kernel<O_GLB_F32><<<BATCH * 2, 128, 0, stream>>>(
      a0, 8, 132, sa3w[0] + 3 * 256, 256, sa3w[0], sa3b[0], 4, 3, 8, 256, a1, 256);
  gemm_kernel<O_GLB_F32><<<BATCH * 4, 128, 0, stream>>>(
      a1, 8, 256, sa3w[1], 512, nullptr, sa3b[1], 8, 0, 8, 512, a2, 512);
  gemm_kernel<O_MAXSTAGE><<<BATCH * 8, 128, 0, stream>>>(
      a2, 8, 512, sa3w[2], 1024, nullptr, sa3b[2], 16, 0, 8, 1024, s3out, 0);

  // ---- Decoder: 1024 -> 512 -> 256 -> 40 (M = 16) ----
  gemm_kernel<O_GLB_F32><<<4, 128, 0, stream>>>(
      s3out, 1, 1024, decw[0], 512, nullptr, decb[0], 32, 0, 8, 512, dact1, 512);
  gemm_kernel<O_GLB_F32><<<2, 128, 0, stream>>>(
      dact1, 1, 512, decw[1], 256, nullptr, decb[1], 16, 0, 8, 256, dact2, 256);
  dec_final_kernel<<<(16 * 40 + 127) / 128, 128, 0, stream>>>(
      dact2, decw[2], decb[2], (float*)d_out);
}
